// DGIR_71717363908902
// MI455X (gfx1250) — compile-verified, matrix operand order checked by reading, not run on hardware
//
#include <hip/hip_runtime.h>
#include <math.h>

typedef __attribute__((ext_vector_type(16))) _Float16 v16h;
typedef __attribute__((ext_vector_type(8)))  float    v8f;
typedef __attribute__((ext_vector_type(2)))  float    floatx2;

union V16H { v16h v; _Float16 h[16]; };
union V8F  { v8f  v; float    f[8];  };

// ============================================================================
// Generic strided WMMA GEMM with optional deterministic split-K:
//   D[z, m, n] = alpha * sum_k A[z,m,k] * B[z,n,k] + bias[n] (+ R[z,m,n])
// Template <AK1,BK1>: operand K-stride == 1 (8B-aligned) -> float2 loads.
// Row out-of-range handling: row index is CLAMPED (garbage rows only affect
// D rows/cols that are never stored), so the main K loop is branch-free.
// Only the K%32 tail zero-selects (v_cndmask, no EXEC divergence).
// ksplit>1: blockIdx.z = z*ksplit+split; each split accumulates its K-range
// into partial tile D + split*pStride (bias/R/alpha must be neutral);
// a separate reduction kernel sums the splits (bit-deterministic, no atomics).
// ============================================================================
template <bool AK1, bool BK1>
__global__ __launch_bounds__(32)
void gemm_wmma_kernel(const float* __restrict__ A, const float* __restrict__ Bm,
                      float* Dm, int M, int N, int K,
                      long sAm, long sAk, long bA,
                      long sBn, long sBk, long bB,
                      long sDm, long sDn, long bD,
                      const float* __restrict__ bias,
                      const float* R, long sRm, long sRn, long bR,
                      float alpha, int ksplit, long pStride)
{
    const int lane  = threadIdx.x & 31;
    const int mloc  = lane & 15;
    const int khalf = lane >> 4;
    const long m0 = (long)blockIdx.x * 16;
    const long n0 = (long)blockIdx.y * 16;

    int zb = blockIdx.z;
    int split = 0;
    long z = zb;
    int kstart = 0, kend = K;
    if (ksplit > 1) {
        split = zb % ksplit;
        z = zb / ksplit;
        int KC = ((K + ksplit * 32 - 1) / (ksplit * 32)) * 32;
        kstart = split * KC;
        if (kstart > K) kstart = K;
        kend = kstart + KC;
        if (kend > K) kend = K;
    }

    long am = m0 + mloc; if (am >= M) am = M - 1;   // clamp: never stored
    long bn = n0 + mloc; if (bn >= N) bn = N - 1;
    const float* Arow = A + z * bA + am * sAm;
    const float* Brow = Bm + z * bB + bn * sBn;

    // per-VGPR K offsets (ISA 7.12.2 f16 A/B fragment layout)
    int kof[8];
#pragma unroll
    for (int v = 0; v < 8; ++v)
        kof[v] = ((v >> 2) << 4) + (khalf << 3) + ((v & 3) << 1);
    long koA[8], koB[8];
#pragma unroll
    for (int v = 0; v < 8; ++v) { koA[v] = (long)kof[v] * sAk; koB[v] = (long)kof[v] * sBk; }

    V8F acc;
#pragma unroll
    for (int i = 0; i < 8; ++i) acc.f[i] = 0.f;

    const int kmain = kstart + ((kend - kstart) & ~31);
    const float* pa = Arow + (long)kstart * sAk;
    const float* pb = Brow + (long)kstart * sBk;

    for (int kb = kstart; kb < kmain; kb += 32) {
        V16H a, b;
#pragma unroll
        for (int v = 0; v < 8; ++v) {
            float a0, a1, b0, b1;
            if (AK1) { floatx2 t = *(const floatx2*)(pa + kof[v]); a0 = t.x; a1 = t.y; }
            else     { const float* q = pa + koA[v]; a0 = q[0]; a1 = q[sAk]; }
            if (BK1) { floatx2 t = *(const floatx2*)(pb + kof[v]); b0 = t.x; b1 = t.y; }
            else     { const float* q = pb + koB[v]; b0 = q[0]; b1 = q[sBk]; }
            a.h[2 * v] = (_Float16)a0;  a.h[2 * v + 1] = (_Float16)a1;
            b.h[2 * v] = (_Float16)b0;  b.h[2 * v + 1] = (_Float16)b1;
        }
        __builtin_prefetch(pa + 32 * sAk, 0, 1);   // global_prefetch_b8
        __builtin_prefetch(pb + 32 * sBk, 0, 1);
        acc.v = __builtin_amdgcn_wmma_f32_16x16x32_f16(
            false, a.v, false, b.v, (short)0, acc.v, false, false);
        pa += 32 * sAk;
        pb += 32 * sBk;
    }

    if (kmain < kend) {   // K tail: clamped address + value select, no branches
        V16H a, b;
#pragma unroll
        for (int v = 0; v < 8; ++v) {
            int k0 = kmain + kof[v], k1 = k0 + 1;
            long o0a = (long)(k0 < K ? k0 : K - 1) * sAk;
            long o1a = (long)(k1 < K ? k1 : K - 1) * sAk;
            long o0b = (long)(k0 < K ? k0 : K - 1) * sBk;
            long o1b = (long)(k1 < K ? k1 : K - 1) * sBk;
            float a0 = Arow[o0a], a1 = Arow[o1a];
            float b0 = Brow[o0b], b1 = Brow[o1b];
            a.h[2 * v]     = (_Float16)(k0 < kend ? a0 : 0.f);
            a.h[2 * v + 1] = (_Float16)(k1 < kend ? a1 : 0.f);
            b.h[2 * v]     = (_Float16)(k0 < kend ? b0 : 0.f);
            b.h[2 * v + 1] = (_Float16)(k1 < kend ? b1 : 0.f);
        }
        acc.v = __builtin_amdgcn_wmma_f32_16x16x32_f16(
            false, a.v, false, b.v, (short)0, acc.v, false, false);
    }

    float* Dp = Dm + (long)split * pStride;
    const long n = n0 + mloc;            // C/D layout: N = lane&15
    if (n < N) {
        const float bs = bias ? bias[n] : 0.f;
#pragma unroll
        for (int v = 0; v < 8; ++v) {
            const long m = m0 + v + (khalf << 3);   // M = vgpr + 8*(lane>=16)
            if (m < M) {
                float r = acc.f[v] * alpha + bs;
                if (R) r += R[z * bR + m * sRm + n * sRn];
                Dp[z * bD + m * sDm + n * sDn] = r;
            }
        }
    }
}

// out[i] = sum_{s<ns} part[s*stride + i]   (deterministic split-K reduction)
__global__ void reduce_sum_kernel(const float* __restrict__ part, float* __restrict__ out,
                                  long n, int ns, long stride)
{
    long idx = (long)blockIdx.x * blockDim.x + threadIdx.x;
    if (idx >= n) return;
    float a = 0.f;
    for (int s = 0; s < ns; ++s) a += part[(long)s * stride + idx];
    out[idx] = a;
}

// ============================================================================
// Direct 3x3 conv, SAME zero padding, NCHW.  w: [Cout,Cin,3,3]
// ============================================================================
__global__ void conv3x3_kernel(const float* __restrict__ x, const float* __restrict__ w,
                               const float* __restrict__ bias, float* __restrict__ y,
                               int Bn, int Cin, int Cout, int H, int W)
{
    long idx = (long)blockIdx.x * blockDim.x + threadIdx.x;
    long total = (long)Bn * Cout * H * W;
    if (idx >= total) return;
    int wv = (int)(idx % W); long t = idx / W;
    int hv = (int)(t % H); t /= H;
    int co = (int)(t % Cout); int b = (int)(t / Cout);
    float acc = bias ? bias[co] : 0.f;
    const float* wb = w + (long)co * Cin * 9;
    const float* xb = x + (long)b * Cin * H * W;
    for (int ci = 0; ci < Cin; ++ci) {
        const float* xc = xb + (long)ci * H * W;
        const float* wc = wb + ci * 9;
        for (int dy = 0; dy < 3; ++dy) {
            int yy = hv + dy - 1;
            if ((unsigned)yy >= (unsigned)H) continue;
            for (int dx = 0; dx < 3; ++dx) {
                int xx = wv + dx - 1;
                if ((unsigned)xx >= (unsigned)W) continue;
                acc += xc[(long)yy * W + xx] * wc[dy * 3 + dx];
            }
        }
    }
    y[idx] = acc;
}

// Depthwise 3x3, SAME.  w: [C,1,3,3]
__global__ void dwconv3x3_kernel(const float* __restrict__ x, const float* __restrict__ w,
                                 float* __restrict__ y, int Bn, int C, int H, int W)
{
    long idx = (long)blockIdx.x * blockDim.x + threadIdx.x;
    long total = (long)Bn * C * H * W;
    if (idx >= total) return;
    int wv = (int)(idx % W); long t = idx / W;
    int hv = (int)(t % H); t /= H;
    int c = (int)(t % C); int b = (int)(t / C);
    const float* xc = x + ((long)b * C + c) * H * W;
    const float* wc = w + (long)c * 9;
    float acc = 0.f;
    for (int dy = 0; dy < 3; ++dy) {
        int yy = hv + dy - 1;
        if ((unsigned)yy >= (unsigned)H) continue;
        for (int dx = 0; dx < 3; ++dx) {
            int xx = wv + dx - 1;
            if ((unsigned)xx >= (unsigned)W) continue;
            acc += xc[(long)yy * W + xx] * wc[dy * 3 + dx];
        }
    }
    y[idx] = acc;
}

// ============================================================================
// Generic LayerNorm over a C-length strided vector; optional per-row mask
// premultiply (mask or 1-mask).  Row (o,i): base = o*so + i*si, elem stride sc.
// ============================================================================
__global__ void layernorm_kernel(const float* __restrict__ x, const float* __restrict__ g,
                                 const float* __restrict__ bt, float* __restrict__ y,
                                 int outer, long inner, long so, long si, long sc,
                                 int C, float eps, const float* __restrict__ mask, int minv)
{
    long idx = (long)blockIdx.x * blockDim.x + threadIdx.x;
    if (idx >= (long)outer * inner) return;
    long o = idx / inner, i = idx % inner;
    const float* xr = x + o * so + i * si;
    float* yr = y + o * so + i * si;
    float mf = 1.f;
    if (mask) { float mm = mask[idx]; mf = minv ? (1.f - mm) : mm; }
    float mu = 0.f;
    for (int c = 0; c < C; ++c) mu += xr[(long)c * sc];
    mu = mu * mf / (float)C;
    float var = 0.f;
    for (int c = 0; c < C; ++c) { float d = xr[(long)c * sc] * mf - mu; var += d * d; }
    var /= (float)C;
    float inv = rsqrtf(var + eps);
    for (int c = 0; c < C; ++c)
        yr[(long)c * sc] = g[c] * ((xr[(long)c * sc] * mf - mu) * inv) + bt[c];
}

// Sum of squares of a contiguous row of length len; row (o,i) base o*so+i*si.
__global__ void rowsumsq_kernel(const float* __restrict__ x, float* __restrict__ out,
                                int outer, long inner, long so, long si, long len)
{
    __shared__ float red[256];
    long row = blockIdx.x;
    long o = row / inner, i = row % inner;
    const float* xr = x + o * so + i * si;
    float s = 0.f;
    for (long t = threadIdx.x; t < len; t += 256) { float v = xr[t]; s += v * v; }
    red[threadIdx.x] = s;
    __syncthreads();
    for (int st = 128; st > 0; st >>= 1) {
        if ((int)threadIdx.x < st) red[threadIdx.x] += red[threadIdx.x + st];
        __syncthreads();
    }
    if (threadIdx.x == 0) out[row] = red[0];
}

// In-place row scale by 1/max(sqrt(ss[row]),1e-12)   (the l2n op)
__global__ void l2scale_kernel(float* __restrict__ x, const float* __restrict__ ss,
                               int outer, long inner, long so, long si, long len)
{
    long idx = (long)blockIdx.x * blockDim.x + threadIdx.x;
    long total = (long)outer * inner * len;
    if (idx >= total) return;
    long l = idx % len; long row = idx / len;
    long o = row / inner, i = row % inner;
    float s = 1.f / fmaxf(sqrtf(ss[row]), 1e-12f);
    x[o * so + i * si + l] *= s;
}

// Row softmax (cols<=128). Optional row multiply temp[(r/rowsPerTemp)%tempMod],
// optional divide by sqrt(rowSS[r]) and sqrt(colSS[c])  (DSAM cosine scaling).
__global__ void softmax_rows_kernel(const float* __restrict__ xin, float* __restrict__ xout,
                                    long rows, int cols,
                                    const float* __restrict__ temp, int rowsPerTemp, int tempMod,
                                    const float* __restrict__ rowSS, const float* __restrict__ colSS)
{
    long r = (long)blockIdx.x * blockDim.x + threadIdx.x;
    if (r >= rows) return;
    float rs = 1.f;
    if (temp)  rs *= temp[(r / rowsPerTemp) % tempMod];
    if (rowSS) rs /= fmaxf(sqrtf(rowSS[r]), 1e-12f);
    float buf[128];
    const float* xr = xin + r * (long)cols;
    float mx = -3.4e38f;
    for (int c = 0; c < cols; ++c) {
        float v = xr[c] * rs;
        if (colSS) v /= sqrtf(colSS[c]);
        buf[c] = v; mx = fmaxf(mx, v);
    }
    float s = 0.f;
    for (int c = 0; c < cols; ++c) { float e = expf(buf[c] - mx); buf[c] = e; s += e; }
    float inv = 1.f / s;
    float* yr = xout + r * (long)cols;
    for (int c = 0; c < cols; ++c) yr[c] = buf[c] * inv;
}

// g[b,c,p] = gelu_exact(f[b,c,p]) * f[b,c+Hc,p]    f: [B,2Hc,HW], g: [B,Hc,HW]
__global__ void gelu_gate_kernel(const float* __restrict__ f, float* __restrict__ g,
                                 int Bn, int Hc, long HW)
{
    long idx = (long)blockIdx.x * blockDim.x + threadIdx.x;
    long total = (long)Bn * Hc * HW;
    if (idx >= total) return;
    long p = idx % HW; long t = idx / HW;
    int c = (int)(t % Hc); int b = (int)(t / Hc);
    const float* base = f + (long)b * (2 * Hc) * HW;
    float y1 = base[(long)c * HW + p];
    float y2 = base[(long)(c + Hc) * HW + p];
    float ge = 0.5f * y1 * (1.f + erff(y1 * 0.70710678118654752f));
    g[idx] = ge * y2;
}

// pixel_unshuffle r=2:  in [B,Cin,2Ho,2Wo] -> out [B,4Cin,Ho,Wo]
__global__ void pixel_unshuffle_kernel(const float* __restrict__ x, float* __restrict__ y,
                                       int Bn, int Cin, int Ho, int Wo)
{
    long idx = (long)blockIdx.x * blockDim.x + threadIdx.x;
    long total = (long)Bn * Cin * 4 * Ho * Wo;
    if (idx >= total) return;
    int wo = (int)(idx % Wo); long t = idx / Wo;
    int ho = (int)(t % Ho); t /= Ho;
    int co = (int)(t % (Cin * 4)); int b = (int)(t / (Cin * 4));
    int c = co >> 2, r1 = (co >> 1) & 1, r2 = co & 1;
    y[idx] = x[(((long)b * Cin + c) * (Ho * 2) + (ho * 2 + r1)) * (long)(Wo * 2) + (wo * 2 + r2)];
}

// [B,C,HW] -> [B,HW,C]
__global__ void nchw_to_nhwc_kernel(const float* __restrict__ x, float* __restrict__ y,
                                    int Bn, int C, long HW)
{
    long idx = (long)blockIdx.x * blockDim.x + threadIdx.x;
    long total = (long)Bn * C * HW;
    if (idx >= total) return;
    int c = (int)(idx % C); long t = idx / C;
    long p = t % HW; int b = (int)(t / HW);
    y[idx] = x[((long)b * C + c) * HW + p];
}

// mv[r] = max_n sim[r,n]^3
__global__ void mv_kernel(const float* __restrict__ sim, float* __restrict__ mv,
                          long rows, int cols)
{
    long r = (long)blockIdx.x * blockDim.x + threadIdx.x;
    if (r >= rows) return;
    const float* sr = sim + r * (long)cols;
    float m = -3.4e38f;
    for (int c = 0; c < cols; ++c) { float v = sr[c]; v = v * v * v; m = fmaxf(m, v); }
    mv[r] = m;
}

__global__ void batch_minmax_kernel(const float* __restrict__ x, float* __restrict__ mm, long per)
{
    __shared__ float smn[256], smx[256];
    long b = blockIdx.x;
    float mn = 3.4e38f, mx = -3.4e38f;
    for (long t = threadIdx.x; t < per; t += 256) {
        float v = x[b * per + t];
        mn = fminf(mn, v); mx = fmaxf(mx, v);
    }
    smn[threadIdx.x] = mn; smx[threadIdx.x] = mx;
    __syncthreads();
    for (int st = 128; st > 0; st >>= 1) {
        if ((int)threadIdx.x < st) {
            smn[threadIdx.x] = fminf(smn[threadIdx.x], smn[threadIdx.x + st]);
            smx[threadIdx.x] = fmaxf(smx[threadIdx.x], smx[threadIdx.x + st]);
        }
        __syncthreads();
    }
    if (threadIdx.x == 0) { mm[2 * b] = smn[0]; mm[2 * b + 1] = smx[0]; }
}

__global__ void mask_kernel(const float* __restrict__ mv, const float* __restrict__ mm,
                            float* __restrict__ mk, long per, long total)
{
    long idx = (long)blockIdx.x * blockDim.x + threadIdx.x;
    if (idx >= total) return;
    long b = idx / per;
    float mn = mm[2 * b], mx = mm[2 * b + 1];
    mk[idx] = 1.f - (mv[idx] - mn) / (mx - mn);
}

// out = x2 + F * beta[c]
__global__ void final_kernel(const float* __restrict__ x2, const float* __restrict__ F,
                             const float* __restrict__ beta, float* __restrict__ out,
                             int Bn, int C, long HW)
{
    long idx = (long)blockIdx.x * blockDim.x + threadIdx.x;
    long total = (long)Bn * C * HW;
    if (idx >= total) return;
    int c = (int)((idx / HW) % C);
    out[idx] = x2[idx] + F[idx] * beta[c];
}

// ============================================================================
// Host-side orchestration
// ============================================================================
static inline dim3 eg256(long n) { return dim3((unsigned)((n + 255) / 256)); }

static void gemm(hipStream_t s, const float* A, const float* Bm, float* D,
                 int M, int N, int K,
                 long sAm, long sAk, long bA,
                 long sBn, long sBk, long bB,
                 long sDm, long sDn, long bD,
                 int batches, const float* bias,
                 const float* R, long sRm, long sRn, long bR, float alpha,
                 int ksplit = 1, long pStride = 0)
{
    dim3 g((unsigned)((M + 15) / 16), (unsigned)((N + 15) / 16),
           (unsigned)(batches * ksplit));
    const bool ak1 = (sAk == 1) && !(sAm & 1) && !(bA & 1);
    const bool bk1 = (sBk == 1) && !(sBn & 1) && !(bB & 1);
#define GEMM_ARGS A, Bm, D, M, N, K, sAm, sAk, bA, sBn, sBk, bB, sDm, sDn, bD, \
                  bias, R, sRm, sRn, bR, alpha, ksplit, pStride
    if (ak1 && bk1)      gemm_wmma_kernel<true,  true ><<<g, 32, 0, s>>>(GEMM_ARGS);
    else if (ak1)        gemm_wmma_kernel<true,  false><<<g, 32, 0, s>>>(GEMM_ARGS);
    else if (bk1)        gemm_wmma_kernel<false, true ><<<g, 32, 0, s>>>(GEMM_ARGS);
    else                 gemm_wmma_kernel<false, false><<<g, 32, 0, s>>>(GEMM_ARGS);
#undef GEMM_ARGS
}

struct BP {
    const float *ln1w, *ln1b, *qkvw, *qkvdw, *temp, *projw, *ln2w, *ln2b, *ffiw, *ffdw, *ffow;
};

static void run_block(hipStream_t s, float* X, float* Y, float* C, float* D,
                      float* Tq, float* Tk, float* Tattn, const BP& p,
                      int Bn, int Cc, int heads, int H, int W, int hdim)
{
    const long HW = (long)H * W;
    const int HWi = (int)HW;
    const int tc = 3 * Cc, ch = Cc / heads, th = 2 * hdim;

    // ln1 (channel LN, NCHW): X -> Y
    layernorm_kernel<<<eg256((long)Bn * HW), 256, 0, s>>>(X, p.ln1w, p.ln1b, Y,
        Bn, HW, (long)Cc * HW, 1L, HW, Cc, 1e-5f, nullptr, 0);
    // qkv 1x1 (GEMM): Y -> C   [B,3C,HW]
    gemm(s, Y, p.qkvw, C, HWi, tc, Cc, 1, HW, (long)Cc * HW, Cc, 1, 0,
         1, HW, (long)tc * HW, Bn, nullptr, nullptr, 0, 0, 0, 1.f);
    // depthwise 3x3: C -> D
    dwconv3x3_kernel<<<eg256((long)Bn * tc * HW), 256, 0, s>>>(C, p.qkvdw, D, Bn, tc, H, W);
    // l2n of q and k rows (per (b,head,chan) over HW)
    rowsumsq_kernel<<<dim3((unsigned)(Bn * Cc)), 256, 0, s>>>(D, Tq, Bn, (long)Cc,
        (long)tc * HW, HW, HW);
    rowsumsq_kernel<<<dim3((unsigned)(Bn * Cc)), 256, 0, s>>>(D + (long)Cc * HW, Tk, Bn,
        (long)Cc, (long)tc * HW, HW, HW);
    l2scale_kernel<<<eg256((long)Bn * Cc * HW), 256, 0, s>>>(D, Tq, Bn, (long)Cc,
        (long)tc * HW, HW, HW);
    l2scale_kernel<<<eg256((long)Bn * Cc * HW), 256, 0, s>>>(D + (long)Cc * HW, Tk, Bn,
        (long)Cc, (long)tc * HW, HW, HW);
    // attn = qn @ kn^T per (b, head)  [ch x ch], K=HW via split-K WMMA.
    // Partials go to C (dead after dwconv); reduced into Tattn. Deterministic.
    const int ksp = (HW >= 65536) ? 64 : 16;
    const long tileSz = (long)heads * ch * ch;
    for (int b = 0; b < Bn; ++b) {
        const float* qb = D + (long)b * tc * HW;
        gemm(s, qb, qb + (long)Cc * HW, C,
             ch, ch, HWi,
             HW, 1, (long)ch * HW,
             HW, 1, (long)ch * HW,
             ch, 1, (long)ch * ch,
             heads, nullptr, nullptr, 0, 0, 0, 1.f,
             ksp, tileSz);
        reduce_sum_kernel<<<eg256(tileSz), 256, 0, s>>>(C, Tattn + b * tileSz,
                                                        tileSz, ksp, tileSz);
    }
    long rows = (long)Bn * heads * ch;
    softmax_rows_kernel<<<eg256(rows), 256, 0, s>>>(Tattn, Tattn, rows, ch,
        p.temp, ch, heads, nullptr, nullptr);
    // out = attn @ v -> Y  [B,C,HW]
    for (int b = 0; b < Bn; ++b) {
        const float* vb = D + ((long)b * tc + 2 * Cc) * HW;
        gemm(s, vb, Tattn + (long)b * heads * ch * ch, Y + (long)b * Cc * HW,
             HWi, ch, ch,
             1, HW, (long)ch * HW,
             ch, 1, (long)ch * ch,
             1, HW, (long)ch * HW,
             heads, nullptr, nullptr, 0, 0, 0, 1.f);
    }
    // proj 1x1 + residual -> X
    gemm(s, Y, p.projw, X, HWi, Cc, Cc, 1, HW, (long)Cc * HW, Cc, 1, 0,
         1, HW, (long)Cc * HW, Bn, nullptr, X, 1, HW, (long)Cc * HW, 1.f);
    // ln2 -> Y
    layernorm_kernel<<<eg256((long)Bn * HW), 256, 0, s>>>(X, p.ln2w, p.ln2b, Y,
        Bn, HW, (long)Cc * HW, 1L, HW, Cc, 1e-5f, nullptr, 0);
    // ffn_in 1x1 -> C  [B,2h,HW]
    gemm(s, Y, p.ffiw, C, HWi, th, Cc, 1, HW, (long)Cc * HW, Cc, 1, 0,
         1, HW, (long)th * HW, Bn, nullptr, nullptr, 0, 0, 0, 1.f);
    // ffn depthwise -> D
    dwconv3x3_kernel<<<eg256((long)Bn * th * HW), 256, 0, s>>>(C, p.ffdw, D, Bn, th, H, W);
    // gelu gate -> C  [B,h,HW]
    gelu_gate_kernel<<<eg256((long)Bn * hdim * HW), 256, 0, s>>>(D, C, Bn, hdim, HW);
    // ffn_out 1x1 + residual -> X
    gemm(s, C, p.ffow, X, HWi, Cc, hdim, 1, HW, (long)hdim * HW, hdim, 1, 0,
         1, HW, (long)Cc * HW, Bn, nullptr, X, 1, HW, (long)Cc * HW, 1.f);
}

extern "C" void kernel_launch(void* const* d_in, const int* in_sizes, int n_in,
                              void* d_out, int out_size, void* d_ws, size_t ws_size,
                              hipStream_t stream)
{
    (void)in_sizes; (void)n_in; (void)out_size; (void)ws_size;
    const float* X0       = (const float*)d_in[0];
    const float* pe_w     = (const float*)d_in[1];
    const float* down_w   = (const float*)d_in[2];
    const float* ds_in_w  = (const float*)d_in[3];
    const float* ds_in_b  = (const float*)d_in[4];
    const float* ds_sem_w = (const float*)d_in[5];
    const float* ds_sem_b = (const float*)d_in[6];
    const float* ds_dr    = (const float*)d_in[7];
    const float* ds_n1_w  = (const float*)d_in[8];
    const float* ds_n1_b  = (const float*)d_in[9];
    const float* ds_n2_w  = (const float*)d_in[10];
    const float* ds_n2_b  = (const float*)d_in[11];
    const float* ds_out_w = (const float*)d_in[12];
    const float* ds_out_b = (const float*)d_in[13];
    const float* dc_nl_w  = (const float*)d_in[14];
    const float* dc_nl_b  = (const float*)d_in[15];
    const float* dc_nr_w  = (const float*)d_in[16];
    const float* dc_nr_b  = (const float*)d_in[17];
    const float* dc_lp1_w = (const float*)d_in[18];
    const float* dc_lp1_b = (const float*)d_in[19];
    const float* dc_rp1_w = (const float*)d_in[20];
    const float* dc_rp1_b = (const float*)d_in[21];
    const float* dc_lp2_w = (const float*)d_in[22];
    const float* dc_lp2_b = (const float*)d_in[23];
    const float* dc_beta  = (const float*)d_in[24];
    BP b1 = { (const float*)d_in[25], (const float*)d_in[26], (const float*)d_in[27],
              (const float*)d_in[28], (const float*)d_in[29], (const float*)d_in[30],
              (const float*)d_in[31], (const float*)d_in[32], (const float*)d_in[33],
              (const float*)d_in[34], (const float*)d_in[35] };
    BP b2 = { (const float*)d_in[36], (const float*)d_in[37], (const float*)d_in[38],
              (const float*)d_in[39], (const float*)d_in[40], (const float*)d_in[41],
              (const float*)d_in[42], (const float*)d_in[43], (const float*)d_in[44],
              (const float*)d_in[45], (const float*)d_in[46] };
    float* out = (float*)d_out;

    const int Bn = 4;
    const int H1 = 256, W1 = 256; const long HW1 = 65536;
    const int H2 = 128, W2 = 128; const long HW2 = 16384;
    const int d1 = 48, d2c = 96, dd = 128, ndr = 64;
    const int h1 = 127, h2 = 255;

    // Workspace arena (fp32 elements)
    float* ws = (float*)d_ws;
    const size_t S1 = (size_t)Bn * d1 * HW1;            // 12,582,912
    const size_t CS = (size_t)Bn * (2 * h1) * HW1;      // 66,584,576 (peak ffn1)
    float* bufX = ws;
    float* bufY = ws + S1;
    float* bufC = ws + 2 * S1;
    float* bufD = bufC + CS;
    float* T    = bufD + CS;
    float* Tq    = T;            // 512
    float* Tk    = T + 512;      // 512
    float* Tattn = T + 1024;     // <= 18432
    float* Tdrn  = T + 20480;    // 8192
    float* Tdl   = T + 28672;    // 65536
    float* Tdr   = T + 94208;    // 64
    float* Tmv   = T + 94336;    // 65536
    float* Tmm   = T + 159872;   // 8
    float* Tmask = T + 160000;   // 65536

    // ---- patch embed: conv3x3 3->48
    conv3x3_kernel<<<eg256((long)Bn * d1 * HW1), 256, 0, stream>>>(
        X0, pe_w, nullptr, bufX, Bn, 3, d1, H1, W1);
    // ---- restormer block 1 (c=48, heads=1)
    run_block(stream, bufX, bufY, bufC, bufD, Tq, Tk, Tattn, b1, Bn, d1, 1, H1, W1, h1);
    // ---- downsample: conv3x3 48->24 then pixel_unshuffle(2) -> x2 [4,96,128,128] in bufX
    conv3x3_kernel<<<eg256((long)Bn * 24 * HW1), 256, 0, stream>>>(
        bufX, down_w, nullptr, bufY, Bn, d1, 24, H1, W1);
    pixel_unshuffle_kernel<<<eg256((long)Bn * 96 * HW2), 256, 0, stream>>>(
        bufY, bufX, Bn, 24, H2, W2);
    // ---- restormer block 2 (c=96, heads=2)
    run_block(stream, bufX, bufY, bufC, bufD, Tq, Tk, Tattn, b2, Bn, d2c, 2, H2, W2, h2);

    // =================== DSAM ===================
    const int HW2i = (int)HW2;
    // ds_in 1x1 + bias -> bufC (xa, NCHW [B,128,HW2])
    gemm(stream, bufX, ds_in_w, bufC, HW2i, dd, d2c, 1, HW2, (long)d2c * HW2,
         d2c, 1, 0, 1, HW2, (long)dd * HW2, Bn, ds_in_b, nullptr, 0, 0, 0, 1.f);
    // ds_sem conv3x3 + bias -> bufD (xs, NCHW)
    conv3x3_kernel<<<eg256((long)Bn * dd * HW2), 256, 0, stream>>>(
        bufC, ds_sem_w, ds_sem_b, bufD, Bn, dd, dd, H2, W2);
    // NCHW -> [B,HW,128] in bufC
    nchw_to_nhwc_kernel<<<eg256((long)Bn * dd * HW2), 256, 0, stream>>>(bufD, bufC, Bn, dd, HW2);
    // xs_n = LN(xs) -> bufD ; xs2 = LN(xs_n) -> bufC  (faithful double norm)
    layernorm_kernel<<<eg256((long)Bn * HW2), 256, 0, stream>>>(bufC, ds_n1_w, ds_n1_b, bufD,
        Bn, HW2, HW2 * (long)dd, (long)dd, 1L, dd, 1e-5f, nullptr, 0);
    layernorm_kernel<<<eg256((long)Bn * HW2), 256, 0, stream>>>(bufD, ds_n1_w, ds_n1_b, bufC,
        Bn, HW2, HW2 * (long)dd, (long)dd, 1L, dd, 1e-5f, nullptr, 0);
    // drn = LN(ds_dr) -> Tdrn [64,128]
    layernorm_kernel<<<eg256((long)ndr), 256, 0, stream>>>(ds_dr, ds_n2_w, ds_n2_b, Tdrn,
        1, (long)ndr, (long)ndr * dd, (long)dd, 1L, dd, 1e-5f, nullptr, 0);
    // c_u = xs2 @ drn^T -> sim [B,HW2,64]
    float* sim = bufD + (size_t)Bn * dd * HW2;
    gemm(stream, bufC, Tdrn, sim, HW2i, ndr, dd, (long)dd, 1, (long)HW2 * dd,
         (long)dd, 1, 0, (long)ndr, 1, (long)HW2 * ndr, Bn, nullptr, nullptr, 0, 0, 0, 1.f);
    // c_dl (from xs_n), c_dr (from drn)
    rowsumsq_kernel<<<dim3((unsigned)(Bn * HW2)), 256, 0, stream>>>(bufD, Tdl, Bn, HW2,
        (long)HW2 * dd, (long)dd, (long)dd);
    rowsumsq_kernel<<<dim3((unsigned)ndr), 256, 0, stream>>>(Tdrn, Tdr, 1, (long)ndr,
        (long)ndr * dd, (long)dd, (long)dd);
    // sim_soft = softmax_n(c_u / (c_dl*c_dr))
    softmax_rows_kernel<<<eg256((long)Bn * HW2), 256, 0, stream>>>(sim, sim,
        (long)Bn * HW2, ndr, nullptr, 1, 1, Tdl, Tdr);
    // degrad = sim_soft @ drn -> bufC [B,HW2,128]
    gemm(stream, sim, Tdrn, bufC, HW2i, dd, ndr, (long)ndr, 1, (long)HW2 * ndr,
         1, (long)dd, 0, (long)dd, 1, (long)HW2 * dd, Bn, nullptr, nullptr, 0, 0, 0, 1.f);
    // degrad96 = ds_out(degrad) + bias -> bufD [B,96,HW2] NCHW
    gemm(stream, bufC, ds_out_w, bufD, HW2i, d2c, dd, (long)dd, 1, (long)HW2 * dd,
         (long)dd, 1, 0, 1, HW2, (long)d2c * HW2, Bn, ds_out_b, nullptr, 0, 0, 0, 1.f);
    // mask from sim^3
    mv_kernel<<<eg256((long)Bn * HW2), 256, 0, stream>>>(sim, Tmv, (long)Bn * HW2, ndr);
    batch_minmax_kernel<<<dim3((unsigned)Bn), 256, 0, stream>>>(Tmv, Tmm, HW2);
    mask_kernel<<<eg256((long)Bn * HW2), 256, 0, stream>>>(Tmv, Tmm, Tmask, HW2, (long)Bn * HW2);

    // =================== DCAM ===================
    const size_t S2 = (size_t)Bn * d2c * HW2;  // 6,291,456
    float* lnL   = bufC;
    float* Ql    = bufC + S2;
    float* lnR   = bufC + 2 * S2;
    float* Qr    = bufC + 3 * S2;
    float* Vv    = bufC + 4 * S2;
    float* Ff    = bufC + 5 * S2;
    float* attnD = bufD + 2 * S2;  // past degrad96; sim region dead by now

    layernorm_kernel<<<eg256((long)Bn * HW2), 256, 0, stream>>>(bufX, dc_nl_w, dc_nl_b, lnL,
        Bn, HW2, (long)d2c * HW2, 1L, HW2, d2c, 1e-6f, Tmask, 0);
    gemm(stream, lnL, dc_lp1_w, Ql, HW2i, d2c, d2c, 1, HW2, (long)d2c * HW2,
         (long)d2c, 1, 0, 1, HW2, (long)d2c * HW2, Bn, dc_lp1_b, nullptr, 0, 0, 0, 1.f);
    layernorm_kernel<<<eg256((long)Bn * HW2), 256, 0, stream>>>(bufD, dc_nr_w, dc_nr_b, lnR,
        Bn, HW2, (long)d2c * HW2, 1L, HW2, d2c, 1e-6f, Tmask, 1);
    gemm(stream, lnR, dc_rp1_w, Qr, HW2i, d2c, d2c, 1, HW2, (long)d2c * HW2,
         (long)d2c, 1, 0, 1, HW2, (long)d2c * HW2, Bn, dc_rp1_b, nullptr, 0, 0, 0, 1.f);
    gemm(stream, bufX, dc_lp2_w, Vv, HW2i, d2c, d2c, 1, HW2, (long)d2c * HW2,
         (long)d2c, 1, 0, 1, HW2, (long)d2c * HW2, Bn, dc_lp2_b, nullptr, 0, 0, 0, 1.f);
    // attn[b,h,w,v] = scale * sum_c Ql[b,c,h,w] Qr[b,c,h,v]   (batch over h per b)
    const float isc = 1.0f / sqrtf((float)d2c);
    for (int b = 0; b < Bn; ++b) {
        gemm(stream, Ql + (size_t)b * d2c * HW2, Qr + (size_t)b * d2c * HW2,
             attnD + (size_t)b * H2 * W2 * W2,
             W2, W2, d2c,
             1, HW2, (long)W2,
             1, HW2, (long)W2,
             (long)W2, 1, (long)W2 * W2,
             H2, nullptr, nullptr, 0, 0, 0, isc);
    }
    softmax_rows_kernel<<<eg256((long)Bn * H2 * W2), 256, 0, stream>>>(attnD, attnD,
        (long)Bn * H2 * W2, W2, nullptr, 1, 1, nullptr, nullptr);
    // F[b,c,h,w] = sum_v P[b,h,w,v] V[b,c,h,v]   (batch over h per b)
    for (int b = 0; b < Bn; ++b) {
        gemm(stream, Vv + (size_t)b * d2c * HW2, attnD + (size_t)b * H2 * W2 * W2,
             Ff + (size_t)b * d2c * HW2,
             d2c, W2, W2,
             HW2, 1, (long)W2,
             (long)W2, 1, (long)W2 * W2,
             HW2, 1, (long)W2,
             H2, nullptr, nullptr, 0, 0, 0, 1.f);
    }
    // out = x2 + F * beta
    final_kernel<<<eg256((long)Bn * d2c * HW2), 256, 0, stream>>>(bufX, Ff, dc_beta, out,
        Bn, d2c, HW2);
}
